// TransformerWithKMeans_43928925504100
// MI455X (gfx1250) — matrix operand order repaired and checked by reading, not compiled, not run on hardware
//
#include <hip/hip_runtime.h>
#include <hip/hip_bf16.h>
#include <math.h>

// ---------------------------------------------------------------------------
// CDNA5 WMMA (wave32): D(16x16 f32) = A(16x32 f16) * B(32x16 f16) + C
// ---------------------------------------------------------------------------
typedef __attribute__((ext_vector_type(16))) _Float16 v16h;
typedef __attribute__((ext_vector_type(8)))  float    v8f;

__device__ __forceinline__ v8f wmma_f16(v16h a, v16h b, v8f c) {
    return __builtin_amdgcn_wmma_f32_16x16x32_f16(false, a, false, b, (short)0, c,
                                                  false, false);
}

__device__ __forceinline__ v16h ld2x8(const _Float16* p0, const _Float16* p1) {
    v16h r;
    uint4* rp = reinterpret_cast<uint4*>(&r);
    rp[0] = *reinterpret_cast<const uint4*>(p0);
    rp[1] = *reinterpret_cast<const uint4*>(p1);
    return r;
}

__device__ __forceinline__ unsigned int packf16(float a, float b) {
    union { _Float16 h[2]; unsigned int u; } t;
    t.h[0] = (_Float16)a;
    t.h[1] = (_Float16)b;
    return t.u;
}

// ---------------------------------------------------------------------------
#define BATCH   8
#define SEQ     1024
#define DMODEL  256
#define NHEAD   8
#define DHEAD   32
#define MFFN    1024
#define ROWS    (BATCH * SEQ)
#define NCLUST  64
#define KMITERS 10

// ---------------------------------------------------------------------------
// Weight convert + transpose: dst[l][n][k] = (f16) src[l][k][n]
// grid = (N, K/256, L), block = 256
// ---------------------------------------------------------------------------
__global__ void cvt_transpose_f16(const float* __restrict__ src,
                                  _Float16* __restrict__ dst, int K, int N) {
    int n = blockIdx.x;
    int k = blockIdx.y * 256 + threadIdx.x;
    int l = blockIdx.z;
    dst[((size_t)l * N + n) * K + k] = (_Float16)src[((size_t)l * K + k) * N + n];
}

// ---------------------------------------------------------------------------
// LayerNorm over D=256, one block per row, f16 output
// ---------------------------------------------------------------------------
__global__ __launch_bounds__(256) void layernorm_f16(
    const float* __restrict__ x, const float* __restrict__ g,
    const float* __restrict__ bb, _Float16* __restrict__ out) {
    int row = blockIdx.x;
    int d   = threadIdx.x;
    float v = x[(size_t)row * DMODEL + d];
    __shared__ float rs[256];
    __shared__ float rq[256];
    rs[d] = v; rq[d] = v * v;
    __syncthreads();
    for (int s = 128; s > 0; s >>= 1) {
        if (d < s) { rs[d] += rs[d + s]; rq[d] += rq[d + s]; }
        __syncthreads();
    }
    float mean = rs[0] * (1.f / DMODEL);
    float var  = rq[0] * (1.f / DMODEL) - mean * mean;
    float inv  = rsqrtf(var + 1e-5f);
    out[(size_t)row * DMODEL + d] = (_Float16)((v - mean) * inv * g[d] + bb[d]);
}

// ---------------------------------------------------------------------------
// WMMA GEMM: C[MxN] = A[MxK](f16 row) * B[KxN], with B given TRANSPOSED as
// BT[NxK] (f16 row).  Block 256 thr = 8 waves, 128x128 tile; each wave does a
// 32x64 subtile (2x4 wmma per 32-K step).  All LDS staging is b128.
// Software-pipelined: next K-panel is loaded into registers while the current
// panel's WMMAs run; panel after next is prefetched (global_prefetch_b8).
// EPI: 0 = scale -> f16 store, 1 = bias+GELU -> f16, 2 = bias+res -> f32.
// M%128==0, N%128==0, K%32==0 (true for all uses here).
// ---------------------------------------------------------------------------
template <int EPI>
__global__ __launch_bounds__(256) void gemm16(
    const _Float16* __restrict__ A, const _Float16* __restrict__ BT,
    void* __restrict__ Cout, const float* __restrict__ bias,
    const float* __restrict__ res, float scale, int M, int N, int K) {
    __shared__ __align__(16) _Float16 As[128][32];   // [m][k]
    __shared__ __align__(16) _Float16 Bs[128][32];   // [n][k]

    const int tid  = threadIdx.x;
    const int wid  = tid >> 5;
    const int lane = tid & 31;
    const int lh   = lane >> 4;
    const int cl   = lane & 15;
    const int wm   = wid >> 1;    // 0..3 -> 32 rows each
    const int wn   = wid & 1;     // 0..1 -> 64 cols each
    const int m0   = blockIdx.y * 128;
    const int n0   = blockIdx.x * 128;

    v8f acc[2][4] = {};

    const int r   = tid >> 1;           // 0..127
    const int seg = (tid & 1) * 16;     // 0 / 16
    const _Float16* aptr = A  + (size_t)(m0 + r) * K + seg;
    const _Float16* bptr = BT + (size_t)(n0 + r) * K + seg;

    // preload first K-panel into registers
    uint4 ra[2], rb[2];
    {
        const uint4* pa = reinterpret_cast<const uint4*>(aptr);
        const uint4* pb = reinterpret_cast<const uint4*>(bptr);
        ra[0] = pa[0]; ra[1] = pa[1];
        rb[0] = pb[0]; rb[1] = pb[1];
    }

    for (int k0 = 0; k0 < K; k0 += 32) {
        {   // commit staged registers to LDS (pure b128)
            uint4* da = reinterpret_cast<uint4*>(&As[r][seg]);
            uint4* db = reinterpret_cast<uint4*>(&Bs[r][seg]);
            da[0] = ra[0]; da[1] = ra[1];
            db[0] = rb[0]; db[1] = rb[1];
        }
        __syncthreads();

        // issue next panel's global loads early (overlap with WMMAs below)
        if (k0 + 32 < K) {
            const uint4* pa = reinterpret_cast<const uint4*>(aptr + k0 + 32);
            const uint4* pb = reinterpret_cast<const uint4*>(bptr + k0 + 32);
            ra[0] = pa[0]; ra[1] = pa[1];
            rb[0] = pb[0]; rb[1] = pb[1];
        }
        // warm caches for the panel after next
        if (k0 + 64 < K) {
            __builtin_prefetch(aptr + k0 + 64, 0, 1);
            __builtin_prefetch(bptr + k0 + 64, 0, 1);
        }

        v16h af[2], bf[4];
#pragma unroll
        for (int mt = 0; mt < 2; ++mt) {
            const _Float16* ap = &As[wm * 32 + mt * 16 + cl][0];
            af[mt] = ld2x8(ap + lh * 16, ap + lh * 16 + 8);
        }
#pragma unroll
        for (int nt = 0; nt < 4; ++nt) {
            const _Float16* bp = &Bs[wn * 64 + nt * 16 + cl][lh * 16];
            bf[nt] = ld2x8(bp, bp + 8);
        }
#pragma unroll
        for (int mt = 0; mt < 2; ++mt)
#pragma unroll
            for (int nt = 0; nt < 4; ++nt)
                acc[mt][nt] = wmma_f16(af[mt], bf[nt], acc[mt][nt]);
        __syncthreads();
    }

#pragma unroll
    for (int mt = 0; mt < 2; ++mt)
#pragma unroll
        for (int nt = 0; nt < 4; ++nt)
#pragma unroll
            for (int j = 0; j < 8; ++j) {
                int rr = m0 + wm * 32 + mt * 16 + lh * 8 + j;
                int cc = n0 + wn * 64 + nt * 16 + cl;
                float v = acc[mt][nt][j];
                if (EPI == 0) {
                    ((_Float16*)Cout)[(size_t)rr * N + cc] = (_Float16)(v * scale);
                } else if (EPI == 1) {
                    v += bias[cc];
                    v = 0.5f * v * (1.0f + erff(v * 0.70710678118654752f));
                    ((_Float16*)Cout)[(size_t)rr * N + cc] = (_Float16)v;
                } else {
                    ((float*)Cout)[(size_t)rr * N + cc] = v + bias[cc] + res[(size_t)rr * N + cc];
                }
            }
}

// ---------------------------------------------------------------------------
// Transpose V: vT[bh][d][n] (f16) from kv16[row][256 + h*32 + d]
// ---------------------------------------------------------------------------
__global__ __launch_bounds__(256) void transpose_v(
    const _Float16* __restrict__ kv16, _Float16* __restrict__ vT) {
    int row = blockIdx.x;            // b*SEQ + n
    int t   = threadIdx.x;           // h*32 + d
    int b = row >> 10, n = row & 1023;
    int h = t >> 5,    d = t & 31;
    vT[(size_t)((b * NHEAD + h) * DHEAD + d) * SEQ + n] =
        kv16[(size_t)row * (2 * DMODEL) + DMODEL + t];
}

// ---------------------------------------------------------------------------
// Flash attention, transposed formulation (no LDS, minimal shuffles).
//   S^T(16k x 16q) = K(16k x 32d) * Q^T(32d x 16q)     [2 wmma / 32 keys]
//   O^T(d x 16q)  += V^T(16d x 32k) * P^T(32k x 16q)   [2 wmma / 32 keys]
// C-layout puts 8 keys per lane for one query -> softmax reduction is
// in-register + one shfl_xor(16).  P^T -> B-fragment needs only an 8-dword
// lane-half swap.  Q is pre-scaled by 1/sqrt(dh) in its GEMM epilogue.
// Block 256 = 8 waves; each wave owns 16 queries.
// ---------------------------------------------------------------------------
__global__ __launch_bounds__(256) void flash_attn(
    const _Float16* __restrict__ q16, const _Float16* __restrict__ kv16,
    const _Float16* __restrict__ vT, _Float16* __restrict__ ao16) {
    const int wid  = threadIdx.x >> 5;
    const int lane = threadIdx.x & 31;
    const int lh   = lane >> 4;
    const int cl   = lane & 15;
    const int bh   = blockIdx.y;                  // b*NHEAD + h
    const int b    = bh >> 3;
    const int h    = bh & 7;
    const int q0   = blockIdx.x * 128 + wid * 16;
    const int qrow = b * SEQ + q0 + cl;

    // Q^T B-fragment: n = query (cl), element e <-> d = lh*16 + e (contiguous)
    const _Float16* qp = q16 + (size_t)qrow * DMODEL + h * DHEAD + lh * 16;
    v16h bq = ld2x8(qp, qp + 8);

    float m = -1e30f, l = 0.f;
    v8f ot0 = {}, ot1 = {};

    for (int kb = 0; kb < SEQ; kb += 32) {
        // K A-fragments: m-row = key, K-dim = d
        const _Float16* kp0 = kv16 + (size_t)(b * SEQ + kb + cl) * (2 * DMODEL) + h * DHEAD;
        const _Float16* kp1 = kv16 + (size_t)(b * SEQ + kb + 16 + cl) * (2 * DMODEL) + h * DHEAD;
        v16h ak0 = ld2x8(kp0 + lh * 16, kp0 + lh * 16 + 8);
        v16h ak1 = ld2x8(kp1 + lh * 16, kp1 + lh * 16 + 8);
        // warm caches for next iteration's K rows (global_prefetch_b8)
        if (kb + 32 < SEQ) {
            __builtin_prefetch(kp0 + 32 * 2 * DMODEL, 0, 1);
            __builtin_prefetch(kp1 + 32 * 2 * DMODEL, 0, 1);
        }
        v8f s0 = {}, s1 = {};
        s0 = wmma_f16(ak0, bq, s0);     // keys kb..kb+15   (x query tile)
        s1 = wmma_f16(ak1, bq, s1);     // keys kb+16..kb+31

        // online softmax over keys for query `cl` (8 keys/lane/tile in regs)
        float pm = -1e30f;
#pragma unroll
        for (int j = 0; j < 8; ++j) pm = fmaxf(pm, fmaxf(s0[j], s1[j]));
        pm = fmaxf(pm, __shfl_xor(pm, 16, 32));
        float nm    = fmaxf(m, pm);
        float alpha = __expf(m - nm);
        float p0[8], p1[8];
        float ps = 0.f;
#pragma unroll
        for (int j = 0; j < 8; ++j) {
            p0[j] = __expf(s0[j] - nm);
            p1[j] = __expf(s1[j] - nm);
            ps += p0[j] + p1[j];
        }
        ps += __shfl_xor(ps, 16, 32);
        l = l * alpha + ps;
        m = nm;
        ot0 *= alpha;
        ot1 *= alpha;

        // build P^T B-fragment: element e <-> key lh*16+e for query cl.
        unsigned int pk0[4], pk1[4], r0[4], r1[4];
#pragma unroll
        for (int i = 0; i < 4; ++i) {
            pk0[i] = packf16(p0[2 * i], p0[2 * i + 1]);
            pk1[i] = packf16(p1[2 * i], p1[2 * i + 1]);
        }
#pragma unroll
        for (int i = 0; i < 4; ++i) {
            r0[i] = __shfl_xor(pk0[i], 16, 32);
            r1[i] = __shfl_xor(pk1[i], 16, 32);
        }
        v16h bp;
        unsigned int* bpu = reinterpret_cast<unsigned int*>(&bp);
#pragma unroll
        for (int i = 0; i < 4; ++i) {
            bpu[i]     = lh ? r1[i]  : pk0[i];   // keys (lh?16:0)+0..7
            bpu[4 + i] = lh ? pk1[i] : r0[i];    // keys (lh?16:0)+8..15
        }

        // V^T A-fragments: m-row = d, K-dim = key (contiguous in vT)
        const _Float16* vp0 = vT + (size_t)(bh * DHEAD + cl) * SEQ + kb;
        const _Float16* vp1 = vT + (size_t)(bh * DHEAD + 16 + cl) * SEQ + kb;
        v16h av0 = ld2x8(vp0 + lh * 16, vp0 + lh * 16 + 8);
        v16h av1 = ld2x8(vp1 + lh * 16, vp1 + lh * 16 + 8);
        ot0 = wmma_f16(av0, bp, ot0);   // d 0..15  x queries
        ot1 = wmma_f16(av1, bp, ot1);   // d 16..31 x queries
    }

    // epilogue: ot0[j] <-> d = lh*8+j, ot1[j] <-> d = 16+lh*8+j, query = cl.
    float inv = 1.f / l;
    unsigned int w0[4], w1[4];
#pragma unroll
    for (int i = 0; i < 4; ++i) {
        w0[i] = packf16(ot0[2 * i] * inv, ot0[2 * i + 1] * inv);
        w1[i] = packf16(ot1[2 * i] * inv, ot1[2 * i + 1] * inv);
    }
    _Float16* outp = ao16 + (size_t)qrow * DMODEL + h * DHEAD;
    *reinterpret_cast<uint4*>(outp + lh * 8)      = *reinterpret_cast<uint4*>(w0);
    *reinterpret_cast<uint4*>(outp + 16 + lh * 8) = *reinterpret_cast<uint4*>(w1);
}

// ---------------------------------------------------------------------------
// K-means (fp32, deterministic)
// ---------------------------------------------------------------------------
__global__ __launch_bounds__(256) void kmeans_assign(
    const float* __restrict__ x, const float* __restrict__ C,
    int* __restrict__ labels) {
    int wid  = threadIdx.x >> 5;
    int lane = threadIdx.x & 31;
    int row  = blockIdx.x * 8 + wid;
    const float* xr = x + (size_t)row * DMODEL;
    const float* c1 = C + (size_t)lane * DMODEL;
    const float* c2 = C + (size_t)(lane + 32) * DMODEL;
    float d1 = 0.f, d2 = 0.f;
    for (int d = 0; d < DMODEL; ++d) {
        float xv = xr[d];
        float a = xv - c1[d];
        float b = xv - c2[d];
        d1 += a * a;
        d2 += b * b;
    }
    float bd = d1; int bi = lane;
    if (d2 < bd) { bd = d2; bi = lane + 32; }
    for (int msk = 16; msk > 0; msk >>= 1) {
        float od = __shfl_xor(bd, msk, 32);
        int   oi = __shfl_xor(bi, msk, 32);
        if (od < bd || (od == bd && oi < bi)) { bd = od; bi = oi; }
    }
    if (lane == 0) labels[row] = bi;
}

__global__ __launch_bounds__(256) void kmeans_update(
    const float* __restrict__ x, const int* __restrict__ labels,
    float* __restrict__ C) {
    int c = blockIdx.x;
    int d = threadIdx.x;
    float s = 0.f, cnt = 0.f;
    for (int i = 0; i < ROWS; ++i) {
        if (labels[i] == c) { s += x[(size_t)i * DMODEL + d]; cnt += 1.f; }
    }
    if (cnt > 0.f) C[(size_t)c * DMODEL + d] = s / cnt;
}

__global__ __launch_bounds__(256) void centers_proj(
    const float* __restrict__ C, const float* __restrict__ kW,
    const float* __restrict__ kb, float* __restrict__ cp) {
    int c = blockIdx.x, dout = threadIdx.x;
    float acc = kb[dout];
    for (int d = 0; d < DMODEL; ++d)
        acc += C[(size_t)c * DMODEL + d] * kW[(size_t)d * DMODEL + dout];
    cp[(size_t)c * DMODEL + dout] = acc;
}

__global__ __launch_bounds__(256) void gather_out(
    const float* __restrict__ cp, const int* __restrict__ labels,
    float* __restrict__ out) {
    int row = blockIdx.x, d = threadIdx.x;
    out[(size_t)row * DMODEL + d] = cp[(size_t)labels[row] * DMODEL + d];
}

// ---------------------------------------------------------------------------
// Host launch
// ---------------------------------------------------------------------------
extern "C" void kernel_launch(void* const* d_in, const int* in_sizes, int n_in,
                              void* d_out, int out_size, void* d_ws, size_t ws_size,
                              hipStream_t stream) {
    (void)in_sizes; (void)n_in; (void)out_size; (void)ws_size;

    const float* in_x  = (const float*)d_in[0];
    const float* ln1_g = (const float*)d_in[1];
    const float* ln1_b = (const float*)d_in[2];
    const float* Wq    = (const float*)d_in[3];
    const float* Wkv   = (const float*)d_in[4];
    const float* Wo    = (const float*)d_in[5];
    const float* bo    = (const float*)d_in[6];
    const float* ln2_g = (const float*)d_in[7];
    const float* ln2_b = (const float*)d_in[8];
    const float* W1    = (const float*)d_in[9];
    const float* b1    = (const float*)d_in[10];
    const float* W2    = (const float*)d_in[11];
    const float* b2    = (const float*)d_in[12];
    const float* kW    = (const float*)d_in[13];
    const float* kb    = (const float*)d_in[14];

    char* base = (char*)d_ws;
    size_t off = 0;
    auto take = [&](size_t bytes) { char* p = base + off; off += (bytes + 255) & ~(size_t)255; return p; };

    float*    xbuf    = (float*)   take((size_t)ROWS * DMODEL * 4);
    _Float16* xn16    = (_Float16*)take((size_t)ROWS * DMODEL * 2);
    _Float16* q16     = (_Float16*)take((size_t)ROWS * DMODEL * 2);
    _Float16* kv16    = (_Float16*)take((size_t)ROWS * 2 * DMODEL * 2);
    _Float16* vTb     = (_Float16*)take((size_t)BATCH * NHEAD * DHEAD * SEQ * 2);
    _Float16* ao16    = (_Float16*)take((size_t)ROWS * DMODEL * 2);
    _Float16* h16     = (_Float16*)take((size_t)ROWS * MFFN * 2);
    _Float16* WqT16   = (_Float16*)take((size_t)4 * DMODEL * DMODEL * 2);
    _Float16* WkvT16  = (_Float16*)take((size_t)4 * DMODEL * 2 * DMODEL * 2);
    _Float16* WoT16   = (_Float16*)take((size_t)4 * DMODEL * DMODEL * 2);
    _Float16* W1T16   = (_Float16*)take((size_t)4 * DMODEL * MFFN * 2);
    _Float16* W2T16   = (_Float16*)take((size_t)4 * MFFN * DMODEL * 2);
    float*    centers = (float*)   take((size_t)NCLUST * DMODEL * 4);
    float*    cproj   = (float*)   take((size_t)NCLUST * DMODEL * 4);
    int*      labels  = (int*)     take((size_t)ROWS * 4);

    hipMemcpyAsync(xbuf, in_x, (size_t)ROWS * DMODEL * 4, hipMemcpyDeviceToDevice, stream);

    // weights -> transposed f16 (BT[n][k]); grid = (N, K/256, L)
    cvt_transpose_f16<<<dim3(DMODEL,     1, 4), 256, 0, stream>>>(Wq,  WqT16,  DMODEL, DMODEL);
    cvt_transpose_f16<<<dim3(2 * DMODEL, 1, 4), 256, 0, stream>>>(Wkv, WkvT16, DMODEL, 2 * DMODEL);
    cvt_transpose_f16<<<dim3(DMODEL,     1, 4), 256, 0, stream>>>(Wo,  WoT16,  DMODEL, DMODEL);
    cvt_transpose_f16<<<dim3(MFFN,       1, 4), 256, 0, stream>>>(W1,  W1T16,  DMODEL, MFFN);
    cvt_transpose_f16<<<dim3(DMODEL,     4, 4), 256, 0, stream>>>(W2,  W2T16,  MFFN,   DMODEL);

    const float qscale = 0.17677669529663687f;   // 1/sqrt(DHEAD)
    const dim3 blk(256);
    for (int l = 0; l < 4; ++l) {
        layernorm_f16<<<ROWS, blk, 0, stream>>>(xbuf, ln1_g + l * DMODEL, ln1_b + l * DMODEL, xn16);
        gemm16<0><<<dim3(DMODEL / 128, ROWS / 128), blk, 0, stream>>>(
            xn16, WqT16 + (size_t)l * DMODEL * DMODEL, q16, nullptr, nullptr,
            qscale, ROWS, DMODEL, DMODEL);
        gemm16<0><<<dim3(2 * DMODEL / 128, ROWS / 128), blk, 0, stream>>>(
            xn16, WkvT16 + (size_t)l * DMODEL * 2 * DMODEL, kv16, nullptr, nullptr,
            1.0f, ROWS, 2 * DMODEL, DMODEL);
        transpose_v<<<ROWS, blk, 0, stream>>>(kv16, vTb);
        flash_attn<<<dim3(SEQ / 128, BATCH * NHEAD), blk, 0, stream>>>(q16, kv16, vTb, ao16);
        gemm16<2><<<dim3(DMODEL / 128, ROWS / 128), blk, 0, stream>>>(
            ao16, WoT16 + (size_t)l * DMODEL * DMODEL, xbuf, bo + l * DMODEL, xbuf,
            1.0f, ROWS, DMODEL, DMODEL);
        layernorm_f16<<<ROWS, blk, 0, stream>>>(xbuf, ln2_g + l * DMODEL, ln2_b + l * DMODEL, xn16);
        gemm16<1><<<dim3(MFFN / 128, ROWS / 128), blk, 0, stream>>>(
            xn16, W1T16 + (size_t)l * DMODEL * MFFN, h16, b1 + l * MFFN, nullptr,
            1.0f, ROWS, MFFN, DMODEL);
        gemm16<2><<<dim3(DMODEL / 128, ROWS / 128), blk, 0, stream>>>(
            h16, W2T16 + (size_t)l * MFFN * DMODEL, xbuf, b2 + l * DMODEL, xbuf,
            1.0f, ROWS, DMODEL, MFFN);
    }

    hipMemcpyAsync(centers, xbuf, (size_t)NCLUST * DMODEL * 4, hipMemcpyDeviceToDevice, stream);
    for (int it = 0; it < KMITERS; ++it) {
        kmeans_assign<<<ROWS / 8, blk, 0, stream>>>(xbuf, centers, labels);
        kmeans_update<<<NCLUST, blk, 0, stream>>>(xbuf, labels, centers);
    }
    kmeans_assign<<<ROWS / 8, blk, 0, stream>>>(xbuf, centers, labels);

    centers_proj<<<NCLUST, blk, 0, stream>>>(centers, kW, kb, cproj);
    gather_out<<<ROWS, blk, 0, stream>>>(cproj, labels, (float*)d_out);
}
